// VimBaseline_13365938225401
// MI455X (gfx1250) — matrix operand order repaired and checked by reading, not compiled, not run on hardware
//
#include <hip/hip_runtime.h>
#include <hip/hip_bf16.h>

#define NUM_CLASSES 2
#define IMG_H 500
#define IMG_W 232
#define EMBED 128
#define D_STATE 16
#define D_INNER 256
#define DT_RANK 8
#define GH 62
#define GW 29
#define L_PATCH 1798
#define TOKEN_POS 899
#define SEQ 1799
#define BATCH 4
#define MTOK (BATCH * SEQ)              /* 7196 */
#define XPROJ_N (DT_RANK + 2 * D_STATE) /* 40 */

typedef __attribute__((ext_vector_type(16))) _Float16 v16h;
typedef __attribute__((ext_vector_type(8)))  _Float16 v8h;
typedef __attribute__((ext_vector_type(8)))  float    v8f;

// ---------------------------------------------------------------------------
// Plain f32 -> f16 conversion (weights, once per call; ~1.9 MB total).
// ---------------------------------------------------------------------------
__global__ void vim_f32_to_f16(const float* __restrict__ src,
                               _Float16* __restrict__ dst, int n)
{
    int i = blockIdx.x * blockDim.x + threadIdx.x;
    if (i < n) dst[i] = (_Float16)src[i];
}

// ---------------------------------------------------------------------------
// Patch embed: conv2d k=8 s=8 on 1 channel == 64-dot per (token, embed),
// cls token inserted at TOKEN_POS, positional embedding added.
// ---------------------------------------------------------------------------
__global__ void vim_patch_embed(const float* __restrict__ x,
                                const float* __restrict__ pw,
                                const float* __restrict__ pb,
                                const float* __restrict__ cls,
                                const float* __restrict__ pos,
                                float* __restrict__ hid)
{
    int idx = blockIdx.x * blockDim.x + threadIdx.x;
    if (idx >= BATCH * SEQ * EMBED) return;
    int e = idx & (EMBED - 1);
    int s = (idx >> 7) % SEQ;
    int b = idx / (SEQ * EMBED);
    float v;
    if (s == TOKEN_POS) {
        v = cls[e];
    } else {
        int l  = (s < TOKEN_POS) ? s : s - 1;
        int gh = l / GW, gw = l % GW;
        const float* xp = x + ((size_t)b * IMG_H + gh * 8) * IMG_W + gw * 8;
        const float* wp = pw + e * 64;
        float acc = pb[e];
#pragma unroll
        for (int ph = 0; ph < 8; ++ph)
#pragma unroll
            for (int pc = 0; pc < 8; ++pc)
                acc += wp[ph * 8 + pc] * xp[ph * IMG_W + pc];
        v = acc;
    }
    hid[idx] = v + pos[s * EMBED + e];
}

// ---------------------------------------------------------------------------
// residual = hidden (first) or residual + hidden; normed16 = f16(rmsnorm*w)
// ---------------------------------------------------------------------------
__global__ __launch_bounds__(128) void vim_add_rmsnorm(
    const float* __restrict__ hid, float* __restrict__ resid,
    const float* __restrict__ w, _Float16* __restrict__ out16, int first)
{
    int t = blockIdx.x;
    int e = threadIdx.x;
    size_t base = (size_t)t * EMBED + e;
    float r = first ? hid[base] : (resid[base] + hid[base]);
    resid[base] = r;
    __shared__ float red[4];
    float s = r * r;
#pragma unroll
    for (int off = 16; off > 0; off >>= 1) s += __shfl_down(s, off, 32);
    if ((e & 31) == 0) red[e >> 5] = s;
    __syncthreads();
    float inv = rsqrtf((red[0] + red[1] + red[2] + red[3]) * (1.0f / EMBED) + 1e-5f);
    out16[base] = (_Float16)(r * inv * w[e]);
}

// ---------------------------------------------------------------------------
// WMMA GEMM: C[M,N] = A[M,K](f16) x W[N,K](f16)^T, both row-major.
// One wave owns a 16x(NT*16) strip; NT is a template constant so the WMMA
// loop has zero runtime guards. Tile math is forced into SGPRs via
// readfirstlane so boundary checks compile to scalar branches (EXEC stays
// all-1s through every WMMA). K in chunks of 32 via v_wmma_f32_16x16x32_f16.
// ---------------------------------------------------------------------------
template <int NT>
__global__ __launch_bounds__(128) void vim_gemm_wmma(
    const _Float16* __restrict__ A, const _Float16* __restrict__ W,
    float* __restrict__ C, int M, int N, int K, int n0base, int ng)
{
    int lane = threadIdx.x & 31;
    int wave = __builtin_amdgcn_readfirstlane(threadIdx.x >> 5);
    int mt = (M + 15) >> 4;
    int t = blockIdx.x * 4 + wave;
    if (t >= mt * ng) return;
    int tm = t / ng;
    int n0 = n0base + (t % ng) * 64;   // full groups are always 64 wide
    int m0 = tm << 4;
    int g  = lane >> 4;
    int cl = lane & 15;

    int arow = m0 + cl; if (arow >= M) arow = M - 1;
    const _Float16* Ap = A + (size_t)arow * K;

    const _Float16* Wp[NT];
#pragma unroll
    for (int j = 0; j < NT; ++j) {
        int wrow = n0 + j * 16 + cl;
        if (wrow >= N) wrow = N - 1;   // partial N tile: clamp, store-guarded
        Wp[j] = W + (size_t)wrow * K;
    }

    v8f acc[NT];
#pragma unroll
    for (int j = 0; j < NT; ++j) acc[j] = (v8f){};

    for (int kk = 0; kk < K; kk += 32) {
        // A fragment, contiguous K order matching B:
        // lanes 0-15: K = kk+[0..15]; lanes 16-31: K = kk+[16..31]
        v16h a = *(const v16h*)(Ap + kk + g * 16);
#pragma unroll
        for (int j = 0; j < NT; ++j) {
            // B fragment: lane group g holds rows K = kk+g*16+[0..15] of W^T
            v16h b = *(const v16h*)(Wp[j] + kk + g * 16);
            acc[j] = __builtin_amdgcn_wmma_f32_16x16x32_f16(
                false, a, false, b, (short)0, acc[j], false, false);
        }
    }

    bool fullM = (m0 + 16 <= M);           // scalar
    bool fullN = (n0 + NT * 16 <= N);      // scalar
    if (fullM && fullN) {
#pragma unroll
        for (int j = 0; j < NT; ++j) {
            int cn = n0 + j * 16 + cl;
#pragma unroll
            for (int r = 0; r < 8; ++r)
                C[(size_t)(m0 + r + 8 * g) * N + cn] = acc[j][r];
        }
    } else {
#pragma unroll
        for (int j = 0; j < NT; ++j) {
            int cn = n0 + j * 16 + cl;
#pragma unroll
            for (int r = 0; r < 8; ++r) {
                int cm = m0 + r + 8 * g;
                if (cm < M && cn < N) C[(size_t)cm * N + cn] = acc[j][r];
            }
        }
    }
}

// ---------------------------------------------------------------------------
// Per-direction causal depthwise conv (k=4) + SiLU; blockIdx.y = direction.
// dir=1 reads the sequence flipped along L. Writes f32 (scan) + f16 (GEMM).
// ---------------------------------------------------------------------------
__global__ void vim_conv_silu(const float* __restrict__ xz,
                              const float* __restrict__ cw0,
                              const float* __restrict__ cb0,
                              const float* __restrict__ cw1,
                              const float* __restrict__ cb1,
                              float* __restrict__ u32o0, _Float16* __restrict__ u16o0,
                              float* __restrict__ u32o1, _Float16* __restrict__ u16o1)
{
    int dir = blockIdx.y;
    int idx = blockIdx.x * blockDim.x + threadIdx.x;
    if (idx >= MTOK * D_INNER) return;
    const float* cw = dir ? cw1 : cw0;
    const float* cb = dir ? cb1 : cb0;
    int d = idx & (D_INNER - 1);
    int j = (idx >> 8) % SEQ;
    int b = idx / (SEQ * D_INNER);
    float acc = cb[d];
#pragma unroll
    for (int k = 0; k < 4; ++k) {
        int jj = j - 3 + k;
        if (jj >= 0) {
            int l = dir ? (SEQ - 1 - jj) : jj;
            acc += xz[((size_t)b * SEQ + l) * (2 * D_INNER) + d] * cw[d * 4 + k];
        }
    }
    float u = acc / (1.0f + __expf(-acc));  // silu
    if (dir) { u32o1[idx] = u; u16o1[idx] = (_Float16)u; }
    else     { u32o0[idx] = u; u16o0[idx] = (_Float16)u; }
}

// ---------------------------------------------------------------------------
// dt = softplus(dtraw @ dt_w^T + dt_b), K = 8 (too small for WMMA; scalar).
// ---------------------------------------------------------------------------
__global__ void vim_dt_softplus(const float* __restrict__ xdbl,
                                const float* __restrict__ dtw,
                                const float* __restrict__ dtb,
                                float* __restrict__ dt)
{
    int idx = blockIdx.x * blockDim.x + threadIdx.x;
    if (idx >= MTOK * D_INNER) return;
    int d = idx & (D_INNER - 1);
    int m = idx >> 8;
    float acc = dtb[d];
#pragma unroll
    for (int r = 0; r < DT_RANK; ++r)
        acc += xdbl[(size_t)m * XPROJ_N + r] * dtw[d * DT_RANK + r];
    dt[idx] = (acc > 20.0f) ? acc : log1pf(__expf(acc));
}

// ---------------------------------------------------------------------------
// Selective scan, both directions concurrently: grid (BATCH, 2).
// Thread d owns h[d][0..15] in registers; per-step B/C broadcast via
// double-buffered LDS (1 barrier/step). Skip (u*D) + SiLU(z) gate fused;
// u/dt streams prefetched 8 steps ahead (global_prefetch_b8).
// ---------------------------------------------------------------------------
__global__ __launch_bounds__(256) void vim_scan2(
    const float* __restrict__ u0, const float* __restrict__ dt0,
    const float* __restrict__ xdbl0, const float* __restrict__ A_log0,
    const float* __restrict__ Dv0,
    const float* __restrict__ u1, const float* __restrict__ dt1,
    const float* __restrict__ xdbl1, const float* __restrict__ A_log1,
    const float* __restrict__ Dv1,
    const float* __restrict__ xz,
    float* __restrict__ y0, float* __restrict__ y1)
{
    int dir = blockIdx.y;
    const float* u     = dir ? u1 : u0;
    const float* dt    = dir ? dt1 : dt0;
    const float* xdbl  = dir ? xdbl1 : xdbl0;
    const float* A_log = dir ? A_log1 : A_log0;
    const float* Dv    = dir ? Dv1 : Dv0;
    float*       y     = dir ? y1 : y0;

    int b = blockIdx.x;
    int d = threadIdx.x;
    float A[D_STATE];
#pragma unroll
    for (int n = 0; n < D_STATE; ++n) A[n] = -__expf(A_log[d * D_STATE + n]);
    float Dd = Dv[d];
    float h[D_STATE];
#pragma unroll
    for (int n = 0; n < D_STATE; ++n) h[n] = 0.0f;

    __shared__ float BC[2][2 * D_STATE];
    size_t rb = (size_t)b * SEQ;
    if (d < 32) BC[0][d] = xdbl[rb * XPROJ_N + DT_RANK + d];  // B: 0..15, C: 16..31

    for (int l = 0; l < SEQ; ++l) {
        __syncthreads();
        if (l + 1 < SEQ && d < 32)
            BC[(l + 1) & 1][d] = xdbl[(rb + l + 1) * XPROJ_N + DT_RANK + d];
        const float* Bs = &BC[l & 1][0];
        const float* Cs = &BC[l & 1][D_STATE];

        size_t row = (rb + l) * D_INNER + d;
        float dtv = dt[row];
        float uv  = u[row];
        float du  = dtv * uv;
        float yv  = 0.0f;
#pragma unroll
        for (int n = 0; n < D_STATE; ++n) {
            h[n] = __expf(dtv * A[n]) * h[n] + du * Bs[n];
            yv  += h[n] * Cs[n];
        }
        yv += uv * Dd;
        int lz = dir ? (SEQ - 1 - l) : l;
        float z = xz[((size_t)b * SEQ + lz) * (2 * D_INNER) + D_INNER + d];
        yv *= z / (1.0f + __expf(-z));     // * silu(z)
        y[row] = yv;
        if (l + 8 < SEQ) {
            __builtin_prefetch(u  + row + 8 * D_INNER, 0, 1);
            __builtin_prefetch(dt + row + 8 * D_INNER, 0, 1);
        }
    }
}

// ---------------------------------------------------------------------------
// ycomb = f16(0.5 * (y_fwd + flip(y_bwd))) for the out_proj WMMA GEMM.
// ---------------------------------------------------------------------------
__global__ void vim_combine(const float* __restrict__ yf,
                            const float* __restrict__ yb,
                            _Float16* __restrict__ out16)
{
    int idx = blockIdx.x * blockDim.x + threadIdx.x;
    if (idx >= MTOK * D_INNER) return;
    int d = idx & (D_INNER - 1);
    int l = (idx >> 8) % SEQ;
    int b = idx / (SEQ * D_INNER);
    float vb = yb[((size_t)b * SEQ + (SEQ - 1 - l)) * D_INNER + d];
    out16[idx] = (_Float16)(0.5f * (yf[idx] + vb));
}

// ---------------------------------------------------------------------------
// Final: residual+hidden at cls token, rmsnorm, linear head -> logits [4,2].
// ---------------------------------------------------------------------------
__global__ __launch_bounds__(128) void vim_head(
    const float* __restrict__ hid, const float* __restrict__ resid,
    const float* __restrict__ nw, const float* __restrict__ hw,
    const float* __restrict__ hb, float* __restrict__ out)
{
    int b = blockIdx.x;
    int e = threadIdx.x;
    size_t base = ((size_t)b * SEQ + TOKEN_POS) * EMBED + e;
    float r = hid[base] + resid[base];
    __shared__ float feat[EMBED];
    __shared__ float red[4];
    float s = r * r;
#pragma unroll
    for (int off = 16; off > 0; off >>= 1) s += __shfl_down(s, off, 32);
    if ((e & 31) == 0) red[e >> 5] = s;
    __syncthreads();
    float inv = rsqrtf((red[0] + red[1] + red[2] + red[3]) * (1.0f / EMBED) + 1e-5f);
    feat[e] = r * inv * nw[e];
    __syncthreads();
    if (e < NUM_CLASSES) {
        float acc = hb[e];
        for (int k = 0; k < EMBED; ++k) acc += feat[k] * hw[e * EMBED + k];
        out[b * NUM_CLASSES + e] = acc;
    }
}

// ---------------------------------------------------------------------------
static inline void launch_gemm(const _Float16* A, const _Float16* W, float* C,
                               int M, int N, int K, hipStream_t s)
{
    int mt  = (M + 15) / 16;
    int nt  = (N + 15) / 16;
    int ngf = nt / 4;        // full 64-wide groups
    int rem = nt % 4;        // remaining 16-wide tiles
    if (ngf > 0) {
        int tiles = mt * ngf;
        vim_gemm_wmma<4><<<(tiles + 3) / 4, 128, 0, s>>>(A, W, C, M, N, K, 0, ngf);
    }
    if (rem) {
        int tiles = mt;
        int n0base = ngf * 64;
        int blocks = (tiles + 3) / 4;
        switch (rem) {
        case 1: vim_gemm_wmma<1><<<blocks, 128, 0, s>>>(A, W, C, M, N, K, n0base, 1); break;
        case 2: vim_gemm_wmma<2><<<blocks, 128, 0, s>>>(A, W, C, M, N, K, n0base, 1); break;
        default: vim_gemm_wmma<3><<<blocks, 128, 0, s>>>(A, W, C, M, N, K, n0base, 1); break;
        }
    }
}

static inline void launch_cvt(const float* src, _Float16* dst, int n, hipStream_t s)
{
    vim_f32_to_f16<<<(n + 255) / 256, 256, 0, s>>>(src, dst, n);
}

extern "C" void kernel_launch(void* const* d_in, const int* in_sizes, int n_in,
                              void* d_out, int out_size, void* d_ws, size_t ws_size,
                              hipStream_t stream)
{
    (void)in_sizes; (void)n_in; (void)out_size; (void)ws_size;
    const float* x        = (const float*)d_in[0];
    const float* patch_w  = (const float*)d_in[1];
    const float* patch_b  = (const float*)d_in[2];
    const float* cls      = (const float*)d_in[3];
    const float* pos      = (const float*)d_in[4];
    const float* norm_f_w = (const float*)d_in[5];
    const float* head_w   = (const float*)d_in[6];
    const float* head_b   = (const float*)d_in[7];

    // workspace carve-up (256B aligned slices)
    char* ws = (char*)d_ws;
    auto carve = [&](size_t bytes) {
        char* p = ws;
        ws += (bytes + 255) & ~(size_t)255;
        return p;
    };
    float*    resid    = (float*)   carve((size_t)MTOK * EMBED * 4);
    float*    hid      = (float*)   carve((size_t)MTOK * EMBED * 4);
    _Float16* normed16 = (_Float16*)carve((size_t)MTOK * EMBED * 2);
    float*    xz       = (float*)   carve((size_t)MTOK * 2 * D_INNER * 4);
    float*    ubuf32[2];  _Float16* ubuf16[2];
    float*    xdbl[2];    float*    dtbuf[2];   float* ybuf[2];
    for (int dir = 0; dir < 2; ++dir) {
        ubuf32[dir] = (float*)   carve((size_t)MTOK * D_INNER * 4);
        ubuf16[dir] = (_Float16*)carve((size_t)MTOK * D_INNER * 2);
        xdbl[dir]   = (float*)   carve((size_t)MTOK * XPROJ_N * 4);
        dtbuf[dir]  = (float*)   carve((size_t)MTOK * D_INNER * 4);
        ybuf[dir]   = (float*)   carve((size_t)MTOK * D_INNER * 4);
    }
    _Float16* ycomb16 = (_Float16*)carve((size_t)MTOK * D_INNER * 2);

    // f16 weight shadows (per layer: in_w, out_w, xproj_w fwd/bwd)
    _Float16* in_w16[8]; _Float16* out_w16[8]; _Float16* xp_w16[8][2];
    for (int i = 0; i < 8; ++i) {
        in_w16[i]    = (_Float16*)carve((size_t)2 * D_INNER * EMBED * 2);
        out_w16[i]   = (_Float16*)carve((size_t)EMBED * D_INNER * 2);
        xp_w16[i][0] = (_Float16*)carve((size_t)XPROJ_N * D_INNER * 2);
        xp_w16[i][1] = (_Float16*)carve((size_t)XPROJ_N * D_INNER * 2);
    }

    // convert all weights to f16 up-front
    for (int i = 0; i < 8; ++i) {
        void* const* L = d_in + 8 + 17 * i;
        launch_cvt((const float*)L[1],  in_w16[i],    2 * D_INNER * EMBED, stream);
        launch_cvt((const float*)L[2],  out_w16[i],   EMBED * D_INNER,     stream);
        launch_cvt((const float*)L[5],  xp_w16[i][0], XPROJ_N * D_INNER,   stream);
        launch_cvt((const float*)L[12], xp_w16[i][1], XPROJ_N * D_INNER,   stream);
    }

    const int nElem = MTOK * EMBED;   // 921088
    vim_patch_embed<<<(nElem + 255) / 256, 256, 0, stream>>>(
        x, patch_w, patch_b, cls, pos, hid);

    const int nDin = MTOK * D_INNER;  // 1842176
    const int gDin = (nDin + 255) / 256;

    for (int layer = 0; layer < 8; ++layer) {
        void* const* L = d_in + 8 + 17 * layer;
        const float* norm_w = (const float*)L[0];

        vim_add_rmsnorm<<<MTOK, 128, 0, stream>>>(hid, resid, norm_w, normed16,
                                                  layer == 0 ? 1 : 0);
        // in_proj: [7196,128] x [128->512]
        launch_gemm(normed16, in_w16[layer], xz, MTOK, 2 * D_INNER, EMBED, stream);

        // both directions: conv+silu
        vim_conv_silu<<<dim3(gDin, 2), 256, 0, stream>>>(
            xz,
            (const float*)L[3], (const float*)L[4],
            (const float*)L[10], (const float*)L[11],
            ubuf32[0], ubuf16[0], ubuf32[1], ubuf16[1]);

        for (int dir = 0; dir < 2; ++dir) {
            // x_proj: [7196,256] x [256->40]
            launch_gemm(ubuf16[dir], xp_w16[layer][dir], xdbl[dir],
                        MTOK, XPROJ_N, D_INNER, stream);
            vim_dt_softplus<<<gDin, 256, 0, stream>>>(
                xdbl[dir], (const float*)L[6 + 7 * dir],
                (const float*)L[7 + 7 * dir], dtbuf[dir]);
        }

        // both directions' scans concurrently: grid (BATCH, 2)
        vim_scan2<<<dim3(BATCH, 2), D_INNER, 0, stream>>>(
            ubuf32[0], dtbuf[0], xdbl[0], (const float*)L[8],  (const float*)L[9],
            ubuf32[1], dtbuf[1], xdbl[1], (const float*)L[15], (const float*)L[16],
            xz, ybuf[0], ybuf[1]);

        vim_combine<<<gDin, 256, 0, stream>>>(ybuf[0], ybuf[1], ycomb16);
        // out_proj: [7196,256] x [256->128] -> new hidden
        launch_gemm(ycomb16, out_w16[layer], hid, MTOK, EMBED, D_INNER, stream);
    }

    vim_head<<<BATCH, 128, 0, stream>>>(hid, resid, norm_f_w, head_w, head_b,
                                        (float*)d_out);
}
